// Attention_87308095193383
// MI455X (gfx1250) — compile-verified, matrix operand order checked by reading, not run on hardware
//
#include <hip/hip_runtime.h>
#include <hip/hip_bf16.h>

#define B_ 4
#define N_ 2048
#define C_ 512
#define H_ 8
#define D_ 64

typedef __attribute__((ext_vector_type(16))) __bf16 v16bf;
typedef __attribute__((ext_vector_type(8)))  __bf16 v8bf;
typedef __attribute__((ext_vector_type(8)))  float  v8f;
typedef __attribute__((ext_vector_type(4)))  int    v4i;

#if defined(__HIP_DEVICE_COMPILE__) && \
    __has_builtin(__builtin_amdgcn_global_load_async_to_lds_b128) && \
    __has_builtin(__builtin_amdgcn_s_wait_asynccnt)
#define ASYNC_OK 1
#else
#define ASYNC_OK 0
#endif

#define AS1 __attribute__((address_space(1)))
#define AS3 __attribute__((address_space(3)))

// 16-byte global -> LDS copy. Async (ASYNCcnt-tracked, no VGPR staging) when
// the gfx1250 async builtins are available; plain vector copy otherwise.
__device__ __forceinline__ void cp16(__bf16* lds, const __bf16* g) {
#if ASYNC_OK
  __builtin_amdgcn_global_load_async_to_lds_b128((AS1 v4i*)g, (AS3 v4i*)lds, 0, 0);
#else
  *(v8bf*)lds = *(const v8bf*)g;
#endif
}
__device__ __forceinline__ void async_wait0() {
#if ASYNC_OK
  __builtin_amdgcn_s_wait_asynccnt(0);
#endif
}

__device__ __forceinline__ __bf16 f2bf(float f) {
  union { float f; unsigned u; } a; a.f = f;
  unsigned r = a.u + 0x7fffu + ((a.u >> 16) & 1u);
  unsigned short h = (unsigned short)(r >> 16);
  __bf16 o; __builtin_memcpy(&o, &h, 2); return o;
}
__device__ __forceinline__ float bf2f(__bf16 b) {
  unsigned short h; __builtin_memcpy(&h, &b, 2);
  union { unsigned u; float f; } a; a.u = ((unsigned)h) << 16; return a.f;
}

// One 16xK fragment row-slice: lane holds k = {ks..ks+7, ks+16..ks+23}
__device__ __forceinline__ v16bf frag16(const __bf16* p, int kstart) {
  v8bf lo = *(const v8bf*)(p + kstart);
  v8bf hi = *(const v8bf*)(p + kstart + 16);
  return __builtin_shufflevector(lo, hi, 0,1,2,3,4,5,6,7,8,9,10,11,12,13,14,15);
}

__device__ __forceinline__ v8f wmma_bf(v16bf a, v16bf b, v8f c) {
  return __builtin_amdgcn_wmma_f32_16x16x32_bf16(false, a, false, b, (short)0, c, false, false);
}

__global__ __launch_bounds__(256)
void k_f2bf(const float* __restrict__ in, __bf16* __restrict__ out, int n) {
  int i = blockIdx.x * 256 + threadIdx.x;
  int stride = gridDim.x * 256;
  for (; i < n; i += stride) out[i] = f2bf(in[i]);
}

#define MODE_F32   0
#define MODE_SIGBF 1
#define MODE_QKV   2

// C[m][n] = sum_k A[m][k] * Bw[n][k]   (A @ B^T, both K-contiguous)
// block tile 128x64, 8 waves (4x2), wave 32x32 = 2x2 wmma tiles, K step 32,
// double-buffered LDS fed by async global->LDS copies.
template<int MODE>
__global__ __launch_bounds__(256)
void k_gemm(const __bf16* __restrict__ A, const __bf16* __restrict__ Bw,
            void* __restrict__ outp, const float* __restrict__ bias,
            int M, int Nc, int K,
            long sAb, long sBb, long sOb,
            __bf16* __restrict__ qO, __bf16* __restrict__ kO, __bf16* __restrict__ vO)
{
  __shared__ __align__(16) __bf16 sA[2][128 * 32];
  __shared__ __align__(16) __bf16 sB[2][64 * 32];
  const int tid = threadIdx.x;
  const int lane = tid & 31, w = tid >> 5;
  const int wm = w & 3, wn = w >> 2;
  const int m0 = blockIdx.y * 128, n0 = blockIdx.x * 64;
  const int z = blockIdx.z;
  A  += (long)z * sAb;
  Bw += (long)z * sBb;

  const int lr  = lane & 15;
  const int kst = (lane >> 4) * 8;

  const int arow = tid >> 1, ahalf = (tid & 1) * 16;   // A: 128 rows x 32 k
  const int brow = tid >> 2, bq = (tid & 3) * 8;       // B: 64 rows x 32 k
  const __bf16* Asrc = A + (long)(m0 + arow) * K + ahalf;
  const __bf16* Bsrc = Bw + (long)(n0 + brow) * K + bq;

  auto fill = [&](int buf, int kb) {
    cp16(&sA[buf][arow * 32 + ahalf],     Asrc + kb);
    cp16(&sA[buf][arow * 32 + ahalf + 8], Asrc + kb + 8);
    cp16(&sB[buf][brow * 32 + bq],        Bsrc + kb);
  };

  v8f acc[2][2] = {};
  fill(0, 0);
  async_wait0();
  __syncthreads();
  int buf = 0;

  for (int kb = 0; kb < K; kb += 32) {
    if (kb + 32 < K) fill(buf ^ 1, kb + 32);   // prefetch next tile (overlaps wmma)
    v16bf af[2], bfg[2];
    af[0]  = frag16(&sA[buf][(wm * 32 + lr) * 32], kst);
    af[1]  = frag16(&sA[buf][(wm * 32 + 16 + lr) * 32], kst);
    bfg[0] = frag16(&sB[buf][(wn * 32 + lr) * 32], kst);
    bfg[1] = frag16(&sB[buf][(wn * 32 + 16 + lr) * 32], kst);
#pragma unroll
    for (int mi = 0; mi < 2; ++mi)
#pragma unroll
      for (int ni = 0; ni < 2; ++ni)
        acc[mi][ni] = wmma_bf(af[mi], bfg[ni], acc[mi][ni]);
    async_wait0();
    __syncthreads();
    buf ^= 1;
  }

  const int rbase = (lane >> 4) * 8;
#pragma unroll
  for (int mi = 0; mi < 2; ++mi)
#pragma unroll
    for (int ni = 0; ni < 2; ++ni) {
      int col = n0 + wn * 32 + ni * 16 + lr;
      float bv = (MODE == MODE_F32 && bias) ? bias[col] : 0.f;
#pragma unroll
      for (int vr = 0; vr < 8; ++vr) {
        int row = m0 + wm * 32 + mi * 16 + rbase + vr;
        float v = acc[mi][ni][vr];
        if (MODE == MODE_F32) {
          ((float*)outp)[(long)z * sOb + (long)row * Nc + col] = v + bv;
        } else if (MODE == MODE_SIGBF) {
          ((__bf16*)outp)[(long)z * sOb + (long)row * Nc + col] =
              f2bf(1.f / (1.f + __expf(-v)));
        } else {  // MODE_QKV: scatter to q / k (row-major) and v^T
          int b = row >> 11, n = row & 2047;
          int which = col >> 9, rem = col & 511;
          int h = rem >> 6, d = rem & 63;
          long bh = (long)(b * H_ + h);
          __bf16 bw = f2bf(v);
          if (which == 0)      qO[(bh * N_ + n) * D_ + d] = bw;
          else if (which == 1) kO[(bh * N_ + n) * D_ + d] = bw;
          else                 vO[(bh * D_ + d) * N_ + n] = bw;
        }
      }
    }
}

// Per (b,group) mean / rstd over 64 channels x 2048 positions (t is [B][N][C])
__global__ __launch_bounds__(256)
void k_gnstats(const float* __restrict__ t, float* __restrict__ stats) {
  int bg = blockIdx.x;          // 0..31
  int b = bg >> 3, g = bg & 7;
  const int cnt = 64 * N_;
  float s = 0.f, sq = 0.f;
  for (int i = threadIdx.x; i < cnt; i += 256) {
    int n = i >> 6, cc = i & 63;
    float v = t[((long)(b * N_ + n)) * C_ + g * 64 + cc];
    s += v; sq += v * v;
  }
  __shared__ float sh1[256], sh2[256];
  sh1[threadIdx.x] = s; sh2[threadIdx.x] = sq;
  __syncthreads();
  for (int o = 128; o > 0; o >>= 1) {
    if (threadIdx.x < o) {
      sh1[threadIdx.x] += sh1[threadIdx.x + o];
      sh2[threadIdx.x] += sh2[threadIdx.x + o];
    }
    __syncthreads();
  }
  if (threadIdx.x == 0) {
    float mean = sh1[0] / cnt;
    float var = sh2[0] / cnt - mean * mean;
    stats[bg * 2]     = mean;
    stats[bg * 2 + 1] = rsqrtf(var + 1e-5f);
  }
}

__global__ __launch_bounds__(256)
void k_gnapply(const float* __restrict__ t, const float* __restrict__ stats,
               const float* __restrict__ gamma, const float* __restrict__ beta,
               __bf16* __restrict__ out) {
  long total = (long)B_ * N_ * C_;
  long stride = (long)gridDim.x * 256;
  for (long i = (long)blockIdx.x * 256 + threadIdx.x; i < total; i += stride) {
    int c = (int)(i & (C_ - 1));
    int b = (int)(i / ((long)N_ * C_));
    int g = c >> 6;
    float mean = stats[(b * 8 + g) * 2], rstd = stats[(b * 8 + g) * 2 + 1];
    out[i] = f2bf((t[i] - mean) * rstd * gamma[c] + beta[c]);
  }
}

// Flash attention: block = 8 waves, each wave owns 16 query rows (128/block).
// S = (Q K^T) * scale * pe_sig; online softmax; O += P V.
// K/V tiles double-buffered via async global->LDS copies.
__global__ __launch_bounds__(256)
void k_flash(const __bf16* __restrict__ q, const __bf16* __restrict__ kk,
             const __bf16* __restrict__ vT, const __bf16* __restrict__ pe,
             __bf16* __restrict__ ao)
{
  __shared__ __align__(16) __bf16 sK[2][64 * 64];
  __shared__ __align__(16) __bf16 sV[2][64 * 64];   // [d][m_local]
  __shared__ __align__(16) __bf16 sP[8 * 16 * 64];
  const int tid = threadIdx.x;
  const int lane = tid & 31, w = tid >> 5;
  const int lr = lane & 15, kst = (lane >> 4) * 8, rbase = (lane >> 4) * 8;
  const int bh = blockIdx.y;
  const int b = bh >> 3, h = bh & 7;
  const int qbase = blockIdx.x * 128 + w * 16;
  const float scale = 0.125f;  // 64^-0.5

  const __bf16* qp = q  + (long)bh * N_ * D_;
  const __bf16* kp = kk + (long)bh * N_ * D_;
  const __bf16* vp = vT + (long)bh * D_ * N_;
  const __bf16* pep = pe + ((long)b * N_ + qbase + rbase) * (long)N_ + lr;

  v16bf qf[2];
  qf[0] = frag16(qp + (long)(qbase + lr) * D_, kst);
  qf[1] = frag16(qp + (long)(qbase + lr) * D_, 32 + kst);

  const int row = tid >> 2, seg = (tid & 3) * 16;   // 64 rows x 64 cols fill
  auto fillkv = [&](int buf, int mt) {
    cp16(&sK[buf][row * 64 + seg],     kp + (long)(mt + row) * D_ + seg);
    cp16(&sK[buf][row * 64 + seg + 8], kp + (long)(mt + row) * D_ + seg + 8);
    cp16(&sV[buf][row * 64 + seg],     vp + (long)row * N_ + mt + seg);
    cp16(&sV[buf][row * 64 + seg + 8], vp + (long)row * N_ + mt + seg + 8);
  };

  v8f oacc[4] = {};
  float mrun[8], lrun[8];
#pragma unroll
  for (int r = 0; r < 8; ++r) { mrun[r] = -1e30f; lrun[r] = 0.f; }

  __bf16* pws = &sP[w * 16 * 64];

  fillkv(0, 0);
  async_wait0();
  __syncthreads();
  int buf = 0;

  for (int mt = 0; mt < N_; mt += 64) {
    if (mt + 64 < N_) fillkv(buf ^ 1, mt + 64);   // prefetch next K/V tile

    float s[4][8];
#pragma unroll
    for (int t = 0; t < 4; ++t) {
      v8f a = {};
      v16bf b0 = frag16(&sK[buf][(t * 16 + lr) * 64], kst);
      v16bf b1 = frag16(&sK[buf][(t * 16 + lr) * 64], 32 + kst);
      a = wmma_bf(qf[0], b0, a);
      a = wmma_bf(qf[1], b1, a);
#pragma unroll
      for (int vr = 0; vr < 8; ++vr) {
        float pev = bf2f(pep[(long)vr * N_ + mt + t * 16]);
        s[t][vr] = a[vr] * scale * pev;
      }
    }
    float corr[8], rsum[8];
#pragma unroll
    for (int vr = 0; vr < 8; ++vr) {
      float mx = fmaxf(fmaxf(s[0][vr], s[1][vr]), fmaxf(s[2][vr], s[3][vr]));
#pragma unroll
      for (int off = 1; off < 16; off <<= 1)
        mx = fmaxf(mx, __shfl_xor(mx, off, 32));
      float mnew = fmaxf(mrun[vr], mx);
      corr[vr] = __expf(mrun[vr] - mnew);
      mrun[vr] = mnew;
      rsum[vr] = 0.f;
    }
#pragma unroll
    for (int t = 0; t < 4; ++t)
#pragma unroll
      for (int vr = 0; vr < 8; ++vr) {
        float p = __expf(s[t][vr] - mrun[vr]);
        s[t][vr] = p;
        rsum[vr] += p;
      }
#pragma unroll
    for (int vr = 0; vr < 8; ++vr) {
      float rs = rsum[vr];
#pragma unroll
      for (int off = 1; off < 16; off <<= 1)
        rs += __shfl_xor(rs, off, 32);
      lrun[vr] = lrun[vr] * corr[vr] + rs;
    }
#pragma unroll
    for (int dt = 0; dt < 4; ++dt)
#pragma unroll
      for (int vr = 0; vr < 8; ++vr)
        oacc[dt][vr] *= corr[vr];
    // stage P (accumulator layout) -> LDS -> A-fragment layout (wave-local, in-order DS)
#pragma unroll
    for (int t = 0; t < 4; ++t)
#pragma unroll
      for (int vr = 0; vr < 8; ++vr)
        pws[(rbase + vr) * 64 + t * 16 + lr] = f2bf(s[t][vr]);
#pragma unroll
    for (int ks = 0; ks < 2; ++ks) {
      v16bf pf = frag16(&pws[lr * 64], ks * 32 + kst);
#pragma unroll
      for (int dt = 0; dt < 4; ++dt) {
        v16bf vf = frag16(&sV[buf][(dt * 16 + lr) * 64], ks * 32 + kst);
        oacc[dt] = wmma_bf(pf, vf, oacc[dt]);
      }
    }
    async_wait0();
    __syncthreads();
    buf ^= 1;
  }
#pragma unroll
  for (int vr = 0; vr < 8; ++vr) {
    float inv = 1.f / lrun[vr];
    long row2 = (long)b * N_ + qbase + rbase + vr;
#pragma unroll
    for (int dt = 0; dt < 4; ++dt)
      ao[row2 * C_ + h * 64 + dt * 16 + lr] = f2bf(oacc[dt][vr] * inv);
  }
}

extern "C" void kernel_launch(void* const* d_in, const int* in_sizes, int n_in,
                              void* d_out, int out_size, void* d_ws, size_t ws_size,
                              hipStream_t stream)
{
  const float* x       = (const float*)d_in[0];
  const float* pe      = (const float*)d_in[1];
  const float* qkv_w   = (const float*)d_in[2];
  const float* proj_w  = (const float*)d_in[3];
  const float* proj_b  = (const float*)d_in[4];
  const float* conv1_w = (const float*)d_in[5];
  const float* conv1_b = (const float*)d_in[6];
  const float* gn1_g   = (const float*)d_in[7];
  const float* gn1_b   = (const float*)d_in[8];
  const float* conv2_w = (const float*)d_in[9];
  const float* conv2_b = (const float*)d_in[10];
  const float* gn2_g   = (const float*)d_in[11];
  const float* gn2_b   = (const float*)d_in[12];

  char* ws = (char*)d_ws;
  size_t off = 0;
  auto alloc = [&](size_t bytes) {
    char* p = ws + off;
    off += (bytes + 255) & ~(size_t)255;
    return p;
  };
  const size_t BNC = (size_t)B_ * N_ * C_;
  __bf16* x_bf   = (__bf16*)alloc(BNC * 2);
  __bf16* pe_bf  = (__bf16*)alloc(BNC * 2);
  __bf16* qw_bf  = (__bf16*)alloc((size_t)3 * C_ * C_ * 2);
  __bf16* w1_bf  = (__bf16*)alloc((size_t)C_ * C_ * 2);
  __bf16* w2_bf  = (__bf16*)alloc((size_t)C_ * C_ * 2);
  __bf16* pw_bf  = (__bf16*)alloc((size_t)C_ * C_ * 2);
  __bf16* q_bf   = (__bf16*)alloc(BNC * 2);
  __bf16* k_bf   = (__bf16*)alloc(BNC * 2);
  __bf16* vT_bf  = (__bf16*)alloc(BNC * 2);
  __bf16* p1_bf  = (__bf16*)alloc(BNC * 2);
  __bf16* p2_bf  = (__bf16*)alloc(BNC * 2);
  float*  t_f    = (float*)alloc(BNC * 4);
  float*  stats  = (float*)alloc(32 * 2 * 4);
  __bf16* pe_sig = (__bf16*)alloc((size_t)B_ * N_ * N_ * 2);
  __bf16* ao_bf  = (__bf16*)alloc(BNC * 2);

  k_f2bf<<<256, 256, 0, stream>>>(x, x_bf, (int)BNC);
  k_f2bf<<<256, 256, 0, stream>>>(pe, pe_bf, (int)BNC);
  k_f2bf<<<64, 256, 0, stream>>>(qkv_w, qw_bf, 3 * C_ * C_);
  k_f2bf<<<32, 256, 0, stream>>>(conv1_w, w1_bf, C_ * C_);
  k_f2bf<<<32, 256, 0, stream>>>(conv2_w, w2_bf, C_ * C_);
  k_f2bf<<<32, 256, 0, stream>>>(proj_w, pw_bf, C_ * C_);

  dim3 blk(256);
  // conv1 -> GN -> p1
  k_gemm<MODE_F32><<<dim3(C_ / 64, (B_ * N_) / 128, 1), blk, 0, stream>>>(
      pe_bf, w1_bf, t_f, conv1_b, B_ * N_, C_, C_, 0, 0, 0, nullptr, nullptr, nullptr);
  k_gnstats<<<32, 256, 0, stream>>>(t_f, stats);
  k_gnapply<<<1024, 256, 0, stream>>>(t_f, stats, gn1_g, gn1_b, p1_bf);
  // conv2 -> GN -> p2   (t_f reused)
  k_gemm<MODE_F32><<<dim3(C_ / 64, (B_ * N_) / 128, 1), blk, 0, stream>>>(
      pe_bf, w2_bf, t_f, conv2_b, B_ * N_, C_, C_, 0, 0, 0, nullptr, nullptr, nullptr);
  k_gnstats<<<32, 256, 0, stream>>>(t_f, stats);
  k_gnapply<<<1024, 256, 0, stream>>>(t_f, stats, gn2_g, gn2_b, p2_bf);
  // qkv projection, scatter to q / k / v^T
  k_gemm<MODE_QKV><<<dim3(3 * C_ / 64, (B_ * N_) / 128, 1), blk, 0, stream>>>(
      x_bf, qw_bf, nullptr, nullptr, B_ * N_, 3 * C_, C_, 0, 0, 0, q_bf, k_bf, vT_bf);
  // pe_attn = sigmoid(p1 @ p2^T) per batch, stored bf16 (L2-resident, shared by all heads)
  k_gemm<MODE_SIGBF><<<dim3(N_ / 64, N_ / 128, B_), blk, 0, stream>>>(
      p1_bf, p2_bf, pe_sig, nullptr, N_, N_, C_,
      (long)N_ * C_, (long)N_ * C_, (long)N_ * N_, nullptr, nullptr, nullptr);
  // fused flash attention
  k_flash<<<dim3(N_ / 128, B_ * H_), blk, 0, stream>>>(q_bf, k_bf, vT_bf, pe_sig, ao_bf);
  // output projection + bias
  k_gemm<MODE_F32><<<dim3(C_ / 64, (B_ * N_) / 128, 1), blk, 0, stream>>>(
      ao_bf, pw_bf, d_out, proj_b, B_ * N_, C_, C_, 0, 0, 0, nullptr, nullptr, nullptr);
}
